// NonogramEnergy_27650999452045
// MI455X (gfx1250) — matrix operand order repaired and checked by reading, not compiled, not run on hardware
//
#include <hip/hip_runtime.h>
#include <hip/hip_bf16.h>
#include <math.h>
#include <stdint.h>

typedef unsigned int   u32;
typedef unsigned short u16;
typedef __attribute__((ext_vector_type(4)))  u32    v4u;
typedef __attribute__((ext_vector_type(16))) __bf16 v16bf;
typedef __attribute__((ext_vector_type(8)))  float  v8f;

#define BATCH   32768
#define DIN     2048
#define DHID    1024
#define CHUNK_K 128
#define LDSP    136   // 128 + 8 pad (bf16 elems); 272B pitch (multiple of 16B)

// round-to-nearest-even f32 -> bf16 (as u32 with result in high bits)
__device__ __forceinline__ u32 rne_bf(float f) {
  u32 u = __float_as_uint(f);
  return u + 0x7FFFu + ((u >> 16) & 1u);
}
// pack two floats to bf16x2 with one v_perm_b32
__device__ __forceinline__ u32 pack2bf(float lo, float hi) {
  return __builtin_amdgcn_perm(rne_bf(hi), rne_bf(lo), 0x07060302u);
}

union Frag { v4u u[2]; v16bf bf; };

// ---------------------------------------------------------------------------
// Kernel 0: one-shot W1 (2048x1024 f32, k-major) -> W1T (1024x2048 bf16,
// n-major). 32x32 tile transpose through LDS; W1T is 4MB -> L2 resident.
// ---------------------------------------------------------------------------
__global__ __launch_bounds__(256)
void NonogramW1TransposeKernel(const float* __restrict__ W1,
                               u16* __restrict__ W1T) {
  __shared__ u16 tile[32][33];
  const int t  = threadIdx.x;
  const int kt = blockIdx.x * 32;   // K tile base
  const int nt = blockIdx.y * 32;   // N tile base

  {
    int r  = t >> 3;          // k within tile
    int c4 = (t & 7) << 2;    // n within tile (x4)
    float4 v = *(const float4*)(W1 + (size_t)(kt + r) * DHID + nt + c4);
    tile[r][c4 + 0] = (u16)(rne_bf(v.x) >> 16);
    tile[r][c4 + 1] = (u16)(rne_bf(v.y) >> 16);
    tile[r][c4 + 2] = (u16)(rne_bf(v.z) >> 16);
    tile[r][c4 + 3] = (u16)(rne_bf(v.w) >> 16);
  }
  __syncthreads();
  {
    int n  = t >> 3;          // n within tile
    int k4 = (t & 7) << 2;    // k within tile (x4)
    u32 lo = (u32)tile[k4 + 0][n] | ((u32)tile[k4 + 1][n] << 16);
    u32 hi = (u32)tile[k4 + 2][n] | ((u32)tile[k4 + 3][n] << 16);
    uint2 pk; pk.x = lo; pk.y = hi;
    *(uint2*)(W1T + (size_t)(nt + n) * DIN + kt + k4) = pk;
  }
}

// --- stage A tile: 64 rows x 128 k, from f32 source (row-major, pitch 1024) ---
__device__ __forceinline__ void stageA_f32(const float* __restrict__ src,
                                           u16* __restrict__ lsA, int t) {
#pragma unroll
  for (int i = 0; i < 8; ++i) {
    int g = t + i * 256;          // 2048 groups of 4
    int r = g >> 5;               // 32 groups per 128-wide row
    int k = (g & 31) << 2;
    float4 v = *(const float4*)(src + r * 1024 + k);
    uint2 pk;
    pk.x = pack2bf(v.x, v.y);
    pk.y = pack2bf(v.z, v.w);
    *(uint2*)(lsA + r * LDSP + k) = pk;
  }
}
// --- stage A tile from int32 source (hint features) ---
__device__ __forceinline__ void stageA_i32(const int* __restrict__ src,
                                           u16* __restrict__ lsA, int t) {
#pragma unroll
  for (int i = 0; i < 8; ++i) {
    int g = t + i * 256;
    int r = g >> 5;
    int k = (g & 31) << 2;
    int4 v = *(const int4*)(src + r * 1024 + k);
    uint2 pk;
    pk.x = pack2bf((float)v.x, (float)v.y);
    pk.y = pack2bf((float)v.z, (float)v.w);
    *(uint2*)(lsA + r * LDSP + k) = pk;
  }
}
// --- stage B tile via CDNA5 async global->LDS DMA (no VGPR round-trip) ---
// lsB[n][k] = W1T[colBlock+n][k0+k]; 64 n x 128 k; 16B per lane per issue.
__device__ __forceinline__ void stageB_async(const u16* __restrict__ srcT,
                                             u16* __restrict__ lsB, int t) {
#pragma unroll
  for (int i = 0; i < 4; ++i) {
    int g  = t + i * 256;         // 1024 groups of 8 bf16
    int n  = g >> 4;              // 16 groups per 128-wide row
    int k8 = (g & 15) << 3;
    const u16* gp = srcT + (size_t)n * DIN + k8;
    u32 lds_off = (u32)(uintptr_t)(lsB + n * LDSP + k8);  // LDS byte offset
    asm volatile("global_load_async_to_lds_b128 %0, %1, off"
                 :: "v"(lds_off), "v"(gp)
                 : "memory");
  }
}
__device__ __forceinline__ void wait_async0() {
  asm volatile("s_wait_asynccnt 0x0" ::: "memory");
}

// --- per-wave MMA over one staged 128-K chunk: two row tiles share one B frag ---
__device__ __forceinline__ void mmaTile(const u16* __restrict__ lsA,
                                        const u16* __restrict__ lsB,
                                        int am0, int am1, int bn, int khalf,
                                        v8f& c0, v8f& c1) {
#pragma unroll
  for (int kk = 0; kk < CHUNK_K; kk += 32) {
    Frag fa0, fa1, fb;
    const u16* pb = lsB + bn * LDSP + kk + khalf * 8;
    fb.u[0] = *(const v4u*)pb;
    fb.u[1] = *(const v4u*)(pb + 16);
    const u16* pa0 = lsA + am0 * LDSP + kk + khalf * 8;
    fa0.u[0] = *(const v4u*)pa0;
    fa0.u[1] = *(const v4u*)(pa0 + 16);
    const u16* pa1 = lsA + am1 * LDSP + kk + khalf * 8;
    fa1.u[0] = *(const v4u*)pa1;
    fa1.u[1] = *(const v4u*)(pa1 + 16);
    c0 = __builtin_amdgcn_wmma_f32_16x16x32_bf16(false, fa0.bf, false, fb.bf,
                                                 (short)0, c0, false, false);
    c1 = __builtin_amdgcn_wmma_f32_16x16x32_bf16(false, fa1.bf, false, fb.bf,
                                                 (short)0, c1, false, false);
  }
}

// ---------------------------------------------------------------------------
// Kernel 1: feats(B x 2048) @ W1(2048 x 1024), fused tanh/b1/W2 epilogue.
// Block = 256 threads = 8 waves (2 row-waves x 4 col-waves); WG tile 64x64.
// blockIdx.x = COLUMN block (fastest) so the 16 col-blocks sharing one A
// panel are adjacent in dispatch -> A streams from HBM once, then L2.
// B staging rides the async-to-LDS unit and overlaps the A-convert VALU work.
// ---------------------------------------------------------------------------
__global__ __launch_bounds__(256)
void NonogramGemmKernel(const float* __restrict__ gridp,
                        const int*   __restrict__ hints,
                        const u16*   __restrict__ W1T,
                        const float* __restrict__ b1,
                        const float* __restrict__ W2,
                        float*       __restrict__ acc) {
  __shared__ u16   lsA[64 * LDSP];
  __shared__ u16   lsB[64 * LDSP];
  __shared__ float lred[256];

  const int t     = threadIdx.x;
  const int lane  = t & 31;
  const int wave  = t >> 5;
  const int wRow  = wave & 1;
  const int wCol  = wave >> 1;
  const int n     = lane & 15;
  const int khalf = lane >> 4;

  const int colBlock = blockIdx.x * 64;   // fastest-varying
  const int rowBlock = blockIdx.y * 64;

  const int am0 = wRow * 16 + n;
  const int am1 = 32 + wRow * 16 + n;
  const int bn  = wCol * 16 + n;

  v8f c0 = {}, c1 = {};

  // Phase 1: grid features, k in [0, 1024)
  for (int k0 = 0; k0 < 1024; k0 += CHUNK_K) {
    stageB_async(W1T + (size_t)colBlock * DIN + k0, lsB, t);  // fire DMA first
    stageA_f32(gridp + (size_t)rowBlock * 1024 + k0, lsA, t); // overlaps DMA
    wait_async0();
    __syncthreads();
    mmaTile(lsA, lsB, am0, am1, bn, khalf, c0, c1);
    __syncthreads();
  }
  // Phase 2: hint features, k in [1024, 2048)
  for (int k0 = 0; k0 < 1024; k0 += CHUNK_K) {
    stageB_async(W1T + (size_t)colBlock * DIN + 1024 + k0, lsB, t);
    stageA_i32(hints + (size_t)rowBlock * 1024 + k0, lsA, t);
    wait_async0();
    __syncthreads();
    mmaTile(lsA, lsB, am0, am1, bn, khalf, c0, c1);
    __syncthreads();
  }

  // --- fused epilogue: sum_rows sum_n tanh(c + b1[n]) * W2[n] ---
  const int col = colBlock + wCol * 16 + n;
  const float b1v = b1[col];
  const float w2v = W2[col];
  float s = 0.f;
#pragma unroll
  for (int i = 0; i < 8; ++i) {
    s += tanhf(c0[i] + b1v) * w2v;
    s += tanhf(c1[i] + b1v) * w2v;
  }

  lred[t] = s;
  __syncthreads();
  for (int off = 128; off; off >>= 1) {
    if (t < off) lred[t] += lred[t + off];
    __syncthreads();
  }
  if (t == 0) atomicAdd(&acc[0], lred[0]);
}

// ---------------------------------------------------------------------------
// Kernel 2: per-board logic + binary penalties. One wave32 per board.
// ---------------------------------------------------------------------------
__global__ __launch_bounds__(256)
void NonogramLogicKernel(const float* __restrict__ gridp,
                         const int*   __restrict__ hints,
                         float*       __restrict__ acc) {
  const int lane = threadIdx.x & 31;
  const int wave = threadIdx.x >> 5;
  const int b    = blockIdx.x * 8 + wave;

  const int* hb = hints + (size_t)b * 1024;   // [2][32][16]
  int rt = 0, ct = 0;
#pragma unroll
  for (int h = 0; h < 16; ++h) {
    rt += hb[lane * 16 + h];
    ct += hb[512 + lane * 16 + h];
  }
  unsigned mrow = __builtin_amdgcn_ballot_w32(rt > 0);
  unsigned mcol = __builtin_amdgcn_ballot_w32(ct > 0);
  int row_last = mrow ? (31 - __clz(mrow)) : -1;
  int col_last = mcol ? (31 - __clz(mcol)) : -1;
  int size = (row_last >= 0 && col_last >= 0)
               ? ((row_last > col_last ? row_last : col_last) + 1) : 12;
  float sizef = (float)size;

  const float* g = gridp + (size_t)b * 1024;
  bool mi = lane < size;

  float ar = 0.f, bsum = 0.f;
  for (int j = 0; j < size; ++j) {
    float gv = g[lane * 32 + j];
    ar   += 1.f / (1.f + expf(-gv * 3.0f));
    bsum += gv * gv;
  }
  float dr = ar - (float)rt;
  float rerr  = mi ? dr * dr : 0.f;
  float bterm = mi ? bsum    : 0.f;

  float ac = 0.f;
  for (int i = 0; i < size; ++i) {
    float gv = g[i * 32 + lane];
    ac += 1.f / (1.f + expf(-gv * 3.0f));
  }
  float dc = ac - (float)ct;
  float cerr = mi ? dc * dc : 0.f;

  float v0 = rerr + cerr, v1 = bterm;
#pragma unroll
  for (int off = 16; off; off >>= 1) {
    v0 += __shfl_xor(v0, off, 32);
    v1 += __shfl_xor(v1, off, 32);
  }
  if (lane == 0) {
    atomicAdd(&acc[1], v0 / sizef);
    atomicAdd(&acc[2], v1 / (sizef * sizef));
  }
}

__global__ void NonogramInitKernel(float* __restrict__ acc) {
  if (threadIdx.x < 3) acc[threadIdx.x] = 0.f;
}

__global__ void NonogramFinalizeKernel(const float* __restrict__ acc,
                                       const float* __restrict__ b2,
                                       float*       __restrict__ out) {
  const float invB = 1.0f / (float)BATCH;
  out[0] = acc[0] * invB + b2[0]
         + acc[1] * (10.0f * invB)
         + acc[2] * (0.1f  * invB);
}

extern "C" void kernel_launch(void* const* d_in, const int* in_sizes, int n_in,
                              void* d_out, int out_size, void* d_ws, size_t ws_size,
                              hipStream_t stream) {
  const float* gridp = (const float*)d_in[0];
  const int*   hints = (const int*)  d_in[1];
  const float* W1    = (const float*)d_in[2];
  const float* b1    = (const float*)d_in[3];
  const float* W2    = (const float*)d_in[4];
  const float* b2    = (const float*)d_in[5];
  float* out = (float*)d_out;

  float* acc = (float*)d_ws;                      // 3 floats
  u16*   W1T = (u16*)((char*)d_ws + 256);         // 1024 x 2048 bf16 (4MB)

  NonogramInitKernel<<<1, 32, 0, stream>>>(acc);

  dim3 tgrid(DIN / 32, DHID / 32);                // 64 x 32 tiles
  NonogramW1TransposeKernel<<<tgrid, 256, 0, stream>>>(W1, W1T);

  dim3 ggrid(DHID / 64, BATCH / 64);              // 16 (cols, fastest) x 512 (rows)
  NonogramGemmKernel<<<ggrid, 256, 0, stream>>>(gridp, hints, W1T, b1, W2, acc);

  NonogramLogicKernel<<<BATCH / 8, 256, 0, stream>>>(gridp, hints, acc);

  NonogramFinalizeKernel<<<1, 1, 0, stream>>>(acc, b2, out);
}
